// ConvCapsuleLayer3D_50835232915491
// MI455X (gfx1250) — compile-verified, matrix operand order checked by reading, not run on hardware
//
#include <hip/hip_runtime.h>
#include <hip/hip_bf16.h>
#include <math.h>

#define NPOS 32768              // 32*32*32 spatial positions
#define NB   4                  // batch
#define NOC  512                // conv output channels = Cin*Cout*Aout
#define SVSTRIDE 66             // padded LDS row stride (dwords) -> no bank conflicts

typedef __attribute__((ext_vector_type(16))) __bf16 v16bf;
typedef __attribute__((ext_vector_type(8)))  float  v8f;

// One lane's half-row of a 64-ci channels-last row: 4 x b128 at element
// offsets kh*8 + {0,16,32,48}.  h[0] = chunk0 (ci 0..31) WMMA fragment,
// h[1] = chunk1 (ci 32..63) fragment, per the ISA 16-bit A/B layout.
union BF16Row { uint4 u[4]; v16bf h[2]; };

// ---------------------------------------------------------------------------
// Kernel 1: input fp32 NCDHW -> bf16 channels-last inT[b][n][ci]
// total elems = 4 * 64 * 32768 = 8388608 (exact grid)
// ---------------------------------------------------------------------------
__global__ __launch_bounds__(256) void pack_input_kernel(
    const float* __restrict__ in, __bf16* __restrict__ inT) {
  int idx = blockIdx.x * 256 + threadIdx.x;
  int n  = idx & (NPOS - 1);
  int ci = (idx >> 15) & 63;
  int b  = idx >> 21;
  inT[(((size_t)b * NPOS + n) << 6) + ci] = (__bf16)in[idx];
}

// ---------------------------------------------------------------------------
// Kernel 2: weights fp32 [oc][ci][tap27] -> bf16 Wp[tap][oc][ci]
// total elems = 512 * 64 * 27 = 884736
// ---------------------------------------------------------------------------
__global__ __launch_bounds__(256) void pack_weights_kernel(
    const float* __restrict__ w, __bf16* __restrict__ wp) {
  int idx = blockIdx.x * 256 + threadIdx.x;
  if (idx >= NOC * 64 * 27) return;
  int tap = idx % 27;
  int ci  = (idx / 27) & 63;
  int oc  = idx / (27 * 64);
  wp[(((size_t)tap * NOC + oc) << 6) + ci] = (__bf16)w[idx];
}

// ---------------------------------------------------------------------------
// Kernel 3: FUSED implicit-GEMM conv (V_WMMA_F32_16X16X32_BF16) + routing.
//
// Workgroup = 16 waves (512 threads) owns one 64-n tile and ALL 512 oc:
//   conv phase:    wave w -> 32oc x 64n tile (8 f32 accumulators),
//                  K-loop = 27 taps x 64 ci, 16 WMMA per tap per wave.
//   votes tile:    parked in LDS (512 oc x 64 n, stride-66 dwords = 132 KB)
//                  -- never touches HBM (saves ~1.1 GB of votes traffic).
//   routing phase: thread (nl, co) = (t&63, t>>6) runs 3 routing iterations
//                  out of LDS; squash/logits decompose per (n, co).
// Grid: 4 b * 512 n-tiles = 2048 blocks.
// ---------------------------------------------------------------------------
__global__ __launch_bounds__(512) void conv_routing_kernel(
    const __bf16* __restrict__ inT, const __bf16* __restrict__ wp,
    const float* __restrict__ bias, float* __restrict__ out) {
  __shared__ float sv[NOC * SVSTRIDE];   // votes tile: sv[oc*66 + n_local]
  __shared__ float sb[128];              // bias (8,16)

  int t    = threadIdx.x;
  int w    = t >> 5;
  int lane = t & 31;
  int nt   = blockIdx.x & 511;           // n-tile (64 positions)
  int b    = blockIdx.x >> 9;

  if (t < 128) sb[t] = bias[t];

  int lm = lane & 15;                    // A: M row / B: N col within 16
  int kh = lane >> 4;                    // K-half select (ISA 16-bit layout)
  int n_base  = nt * 64;
  int oc_base = w * 32;                  // 16 waves cover all 512 oc

  // per-n-subtile unshifted coordinates (hoisted out of the tap loop)
  int x0[4], y0[4], z0[4];
#pragma unroll
  for (int f = 0; f < 4; ++f) {
    int n = n_base + f * 16 + lm;
    x0[f] = n & 31;
    y0[f] = (n >> 5) & 31;
    z0[f] = n >> 10;
  }

  const __bf16* ibase = inT + (((size_t)b * NPOS) << 6) + kh * 8;

  v8f acc[4][2];
#pragma unroll
  for (int f = 0; f < 4; ++f) {
    acc[f][0] = v8f{};
    acc[f][1] = v8f{};
  }

  // ---------------- conv phase: K-loop over 27 taps x 64 ci ----------------
  for (int tap = 0; tap < 27; ++tap) {
    int dz = tap / 9 - 1;
    int dy = (tap / 3) % 3 - 1;
    int dx = tap % 3 - 1;

    // A fragments: both ci-chunks for the wave's two 16-oc subtiles
    BF16Row a0, a1;
    {
      const __bf16* ap0 =
          wp + (((size_t)(tap * NOC + oc_base + lm)) << 6) + kh * 8;
      const __bf16* ap1 = ap0 + (16 << 6);
#pragma unroll
      for (int q = 0; q < 4; ++q) {
        a0.u[q] = *(const uint4*)(ap0 + q * 16);
        a1.u[q] = *(const uint4*)(ap1 + q * 16);
      }
      if (tap < 26)  // lookahead on next tap's weight slice
        __builtin_prefetch(ap0 + ((size_t)NOC << 6), 0, 3);
    }

#pragma unroll
    for (int f = 0; f < 4; ++f) {
      int x = x0[f] + dx;
      int y = y0[f] + dy;
      int z = z0[f] + dz;
      BF16Row bm;
      if (((unsigned)x < 32u) && ((unsigned)y < 32u) && ((unsigned)z < 32u)) {
        int nsh = (z << 10) + (y << 5) + x;
        const __bf16* bp = ibase + (((size_t)nsh) << 6);
#pragma unroll
        for (int q = 0; q < 4; ++q) bm.u[q] = *(const uint4*)(bp + q * 16);
      } else {  // halo -> zero registers (keep EXEC all-1s for WMMA)
#pragma unroll
        for (int q = 0; q < 4; ++q) bm.u[q] = make_uint4(0u, 0u, 0u, 0u);
      }
      acc[f][0] = __builtin_amdgcn_wmma_f32_16x16x32_bf16(
          false, a0.h[0], false, bm.h[0], (short)0, acc[f][0], false, false);
      acc[f][0] = __builtin_amdgcn_wmma_f32_16x16x32_bf16(
          false, a0.h[1], false, bm.h[1], (short)0, acc[f][0], false, false);
      acc[f][1] = __builtin_amdgcn_wmma_f32_16x16x32_bf16(
          false, a1.h[0], false, bm.h[0], (short)0, acc[f][1], false, false);
      acc[f][1] = __builtin_amdgcn_wmma_f32_16x16x32_bf16(
          false, a1.h[1], false, bm.h[1], (short)0, acc[f][1], false, false);
    }
  }

  // park votes tile in LDS.  C layout: VGPR v -> row M = v + 8*kh, col N = lm
  // banks: lanes 0-15 -> b..b+15, lanes 16-31 (+8*66 dwords = +16 banks) ->
  // b+16..b+31: conflict-free.
#pragma unroll
  for (int f = 0; f < 4; ++f) {
#pragma unroll
    for (int s = 0; s < 2; ++s) {
#pragma unroll
      for (int v = 0; v < 8; ++v) {
        int oc = oc_base + s * 16 + v + kh * 8;
        int nl = f * 16 + lm;
        sv[oc * SVSTRIDE + nl] = acc[f][s][v];
      }
    }
  }
  __syncthreads();

  // ---------------- routing phase: thread (nl, co), all data in LDS --------
  {
    int nl = t & 63;
    int co = t >> 6;

    float logit[4];
#pragma unroll
    for (int ci = 0; ci < 4; ++ci) logit[ci] = 1.0f;

#pragma unroll
    for (int it = 0; it < 3; ++it) {
      float route[4];
#pragma unroll
      for (int ci = 0; ci < 4; ++ci)
        route[ci] = 1.0f / (1.0f + __expf(-logit[ci]));

      float vv[4][16];
#pragma unroll
      for (int ci = 0; ci < 4; ++ci)
#pragma unroll
        for (int ao = 0; ao < 16; ++ao)
          vv[ci][ao] = sv[(ci * 128 + co * 16 + ao) * SVSTRIDE + nl];

      float act[16];
      float sq = 0.0f;
#pragma unroll
      for (int ao = 0; ao < 16; ++ao) {
        float pa = sb[co * 16 + ao];
#pragma unroll
        for (int ci = 0; ci < 4; ++ci) pa += route[ci] * vv[ci][ao];
        act[ao] = pa;
        sq += pa * pa;
      }
      float scale = sq / ((1.0f + sq) * sqrtf(sq + 1e-8f));
#pragma unroll
      for (int ao = 0; ao < 16; ++ao) act[ao] *= scale;

      if (it < 2) {
#pragma unroll
        for (int ci = 0; ci < 4; ++ci) {
          float d = 0.0f;
#pragma unroll
          for (int ao = 0; ao < 16; ++ao) d += vv[ci][ao] * act[ao];
          logit[ci] += d;
        }
      } else {
        // out[((b*8+co)*16+ao)*NPOS + n] : nl lane-consecutive -> coalesced
#pragma unroll
        for (int ao = 0; ao < 16; ++ao)
          out[(((size_t)(b * 128 + co * 16 + ao)) << 15) + n_base + nl] =
              act[ao];
      }
    }
  }
}

// ---------------------------------------------------------------------------
extern "C" void kernel_launch(void* const* d_in, const int* in_sizes, int n_in,
                              void* d_out, int out_size, void* d_ws, size_t ws_size,
                              hipStream_t stream) {
  const float* input  = (const float*)d_in[0];   // (4,4,16,32,32,32) fp32
  const float* conv_w = (const float*)d_in[1];   // (512,64,3,3,3)    fp32
  const float* bias   = (const float*)d_in[2];   // (8,16)            fp32
  float* out = (float*)d_out;                    // (4,8,16,32,32,32) fp32

  char* ws = (char*)d_ws;
  const size_t inT_bytes = (size_t)NB * NPOS * 64 * sizeof(__bf16); // 16.8 MB
  __bf16* inT = (__bf16*)ws;
  __bf16* wp  = (__bf16*)(ws + inT_bytes);                          // 1.8 MB

  pack_input_kernel<<<(NB * 64 * NPOS) / 256, 256, 0, stream>>>(input, inT);
  pack_weights_kernel<<<(NOC * 64 * 27 + 255) / 256, 256, 0, stream>>>(conv_w, wp);
  conv_routing_kernel<<<NB * 512, 512, 0, stream>>>(inT, wp, bias, out);
}
